// SparseAttention_65687229825506
// MI455X (gfx1250) — matrix operand order repaired and checked by reading, not compiled, not run on hardware
//
#include <hip/hip_runtime.h>
#include <hip/hip_bf16.h>

// ---------------------------------------------------------------------------
// CDNA5 (gfx1250, wave32) WMMA types: D(16x16 f32) = A(16x32 bf16)*B(32x16 bf16)+C
// ---------------------------------------------------------------------------
typedef __attribute__((ext_vector_type(16))) __bf16 v16bf;
typedef __attribute__((ext_vector_type(8)))  float  v8f;
typedef __attribute__((ext_vector_type(4)))  unsigned int uint32x4;
typedef __attribute__((ext_vector_type(8)))  unsigned int uint32x8;

union BFrag {
    v16bf v;
    unsigned short u[16];
    uint4 q[2];
};

__device__ __forceinline__ unsigned short f2bf(float f) {
    union { float f; unsigned int u; } x;
    x.f = f;
    unsigned int u = x.u;
    unsigned int r = u + 0x7FFFu + ((u >> 16) & 1u);   // round-to-nearest-even
    return (unsigned short)(r >> 16);
}

__device__ __forceinline__ v8f wmma_bf16(const BFrag& a, const BFrag& b, v8f c) {
    return __builtin_amdgcn_wmma_f32_16x16x32_bf16(
        false, a.v, false, b.v, (short)0, c, false, false);
}

// ---- CDNA5 async global->LDS copy (ASYNCcnt) --------------------------------
// GVS mode: mem_addr = SGPR_base + VGPR_byte_offset ; LDS dest = VGPR lds offset
__device__ __forceinline__ void async_ld_b128(unsigned lds_off, unsigned goff,
                                              const void* base) {
    asm volatile("global_load_async_to_lds_b128 %0, %1, %2"
                 :: "v"(lds_off), "v"(goff), "s"(base) : "memory");
}
__device__ __forceinline__ void wait_async0() {
    asm volatile("s_wait_asynccnt 0x0" ::: "memory");
}
// ---- CDNA5 Tensor Data Mover 2D tile load (TENSORcnt) -----------------------
__device__ __forceinline__ void tdm_load_2d(uint32x4 g0, uint32x8 g1) {
    asm volatile("tensor_load_to_lds %0, %1" :: "s"(g0), "s"(g1) : "memory");
}
__device__ __forceinline__ void wait_tensor0() {
    asm volatile("s_wait_tensorcnt 0x0" ::: "memory");
}

__device__ __forceinline__ unsigned lds_off_of(const void* p) {
    return (unsigned)(size_t)p;   // generic LDS pointer: low 32 bits = LDS byte addr
}

// ---------------------------------------------------------------------------
// Elementwise f32 -> bf16 (8 elems/thread, 16B stores)
// ---------------------------------------------------------------------------
__global__ __launch_bounds__(256)
void cvt_f32_bf16(const float* __restrict__ src, unsigned short* __restrict__ dst,
                  int n8) {
    int i = (int)blockIdx.x * 256 + (int)threadIdx.x;
    if (i >= n8) return;
    const float4* s = (const float4*)src + (size_t)i * 2;
    float4 a = s[0], b = s[1];
    union { uint4 q; unsigned short u[8]; } o;
    o.u[0] = f2bf(a.x); o.u[1] = f2bf(a.y); o.u[2] = f2bf(a.z); o.u[3] = f2bf(a.w);
    o.u[4] = f2bf(b.x); o.u[5] = f2bf(b.y); o.u[6] = f2bf(b.z); o.u[7] = f2bf(b.w);
    ((uint4*)dst)[i] = o.q;
}

// ---------------------------------------------------------------------------
// GEMM: C[M,N] = A[M,K](bf16) * W[N,K](bf16)^T, C = f32 or bf16.
// Block tile 128x128, BK=32, 256 threads = 8 waves as 4(M)x2(N);
// wave tile 32x64 = 2 A-frags x 4 B-frags = 8 WMMA per K-step.
// A tile staged via per-lane async b128 copies; W tile via one TDM descriptor.
// ---------------------------------------------------------------------------
#define GBM 128
#define GBN 128
#define GBK 32

template <bool BF16OUT>
__global__ __launch_bounds__(256)
void gemm_bf16(const unsigned short* __restrict__ A,
               const unsigned short* __restrict__ W,
               void* __restrict__ Cout, int M, int N, int K) {
    __shared__ __attribute__((aligned(16))) unsigned short lA[GBM * GBK]; // 8KB
    __shared__ __attribute__((aligned(16))) unsigned short lB[GBN * GBK]; // 8KB

    const int tid  = threadIdx.x;
    const int lane = tid & 31;
    const int wave = tid >> 5;
    const int wm   = wave & 3;   // 32 rows each
    const int wn   = wave >> 2;  // 64 cols each
    const int bm   = (int)blockIdx.x * GBM;
    const int bn   = (int)blockIdx.y * GBN;

    const v8f vzero = {0.f, 0.f, 0.f, 0.f, 0.f, 0.f, 0.f, 0.f};
    v8f acc[2][4];
#pragma unroll
    for (int s = 0; s < 2; ++s)
#pragma unroll
        for (int f = 0; f < 4; ++f) acc[s][f] = vzero;

    const unsigned ldsA = lds_off_of(&lA[0]);
    const unsigned ldsB = lds_off_of(&lB[0]);

    for (int k0 = 0; k0 < K; k0 += GBK) {
        // ---- A tile: 128 rows x 32 bf16 (64B/row) = 512 x 16B chunks, 2/thread
#pragma unroll
        for (int c = 0; c < 2; ++c) {
            int ch = tid * 2 + c;
            int r  = ch >> 2;
            int co = (ch & 3) * 8;
            unsigned goff = (unsigned)(((size_t)(bm + r) * K + k0 + co) * 2);
            async_ld_b128(ldsA + (unsigned)(r * GBK + co) * 2, goff, A);
        }
        // ---- W tile: one TDM 2D descriptor, tile 32(K) x 128(N), data_size=2B
        if (wave == 0) {
            size_t gaddr = (size_t)W + ((size_t)bn * K + k0) * 2;
            uint32x4 g0;
            g0[0] = 1u;                                   // count=1, user mode
            g0[1] = ldsB;                                 // lds_addr
            g0[2] = (unsigned)(gaddr & 0xFFFFFFFFu);      // global_addr lo
            g0[3] = (unsigned)((gaddr >> 32) & 0x01FFFFFFu) | (2u << 30); // type=2
            uint32x8 g1;
            g1[0] = (1u << 16);                           // data_size=2B
            g1[1] = ((unsigned)K & 0xFFFFu) << 16;        // tensor_dim0 lo16
            g1[2] = ((unsigned)K >> 16) | (((unsigned)N & 0xFFFFu) << 16);
            g1[3] = ((unsigned)N >> 16) | (32u << 16);    // tile_dim0 = 32
            g1[4] = 128u;                                 // tile_dim1 = 128 rows
            g1[5] = (unsigned)K;                          // tensor_dim0_stride lo
            g1[6] = 0u;
            g1[7] = 0u;
            tdm_load_2d(g0, g1);
            wait_tensor0();
        }
        wait_async0();
        __syncthreads();

        // ---- fragments + 8 WMMAs ----
        const int kb = (lane < 16) ? 0 : 8;   // A-operand K sub-base
        const int kk = (lane < 16) ? 0 : 8;   // B-operand K sub-base
        BFrag a[2];
#pragma unroll
        for (int s = 0; s < 2; ++s) {
            int row = wm * 32 + s * 16 + (lane & 15);
            a[s].q[0] = *(const uint4*)&lA[row * GBK + kb];
            a[s].q[1] = *(const uint4*)&lA[row * GBK + kb + 16];
        }
#pragma unroll
        for (int f = 0; f < 4; ++f) {
            BFrag b;
            int n = wn * 64 + f * 16 + (lane & 15);
            b.q[0] = *(const uint4*)&lB[n * GBK + kk];
            b.q[1] = *(const uint4*)&lB[n * GBK + kk + 16];
            acc[0][f] = wmma_bf16(a[0], b, acc[0][f]);
            acc[1][f] = wmma_bf16(a[1], b, acc[1][f]);
        }
        __syncthreads();
    }

    // ---- store: C layout (v,lane) -> row v + 8*(lane>=16), col lane%16 ----
    const int cbase = bn + wn * 64 + (lane & 15);
#pragma unroll
    for (int s = 0; s < 2; ++s) {
        int rbase = bm + wm * 32 + s * 16 + ((lane & 16) ? 8 : 0);
#pragma unroll
        for (int f = 0; f < 4; ++f)
#pragma unroll
            for (int v = 0; v < 8; ++v) {
                size_t idx = (size_t)(rbase + v) * N + cbase + f * 16;
                if (BF16OUT) ((unsigned short*)Cout)[idx] = f2bf(acc[s][f][v]);
                else         ((float*)Cout)[idx] = acc[s][f][v];
            }
    }
}

// ---------------------------------------------------------------------------
// Sparse attention: mask(r,c) = (c<=r) && (r-c<=128 || c%64==0)
// One wave per (b, h, 16-query-row tile). Flash-style softmax over 32-key
// chunks: local chunks (c<=r && r-c<=128) + one gathered strided chunk of
// keys c=64*j (c<=r && r-c>128) -- disjoint conditions => exact union.
// qkv is bf16; K/V chunks staged to LDS with async b128 copies.
// ---------------------------------------------------------------------------
#define ATT_T   2048
#define ATT_H   16
#define ATT_HD  64
#define ATT_C   1024
#define ATT_C3  3072

__global__ __launch_bounds__(32)
void sparse_attn(const unsigned short* __restrict__ qkv,
                 unsigned short* __restrict__ out) {
    const int id   = (int)blockIdx.x;
    const int QT   = ATT_T / 16;
    const int qt   = id % QT;
    const int h    = (id / QT) % ATT_H;
    const int b    = id / (QT * ATT_H);
    const int r0   = qt * 16;
    const int lane = (int)threadIdx.x;

    __shared__ __attribute__((aligned(16))) unsigned short lK[32 * ATT_HD]; // [slot][d]
    __shared__ __attribute__((aligned(16))) unsigned short lV[32 * ATT_HD]; // [slot][d]
    __shared__ __attribute__((aligned(16))) unsigned short lP[16 * 32];     // [row][slot]

    const v8f vzero = {0.f, 0.f, 0.f, 0.f, 0.f, 0.f, 0.f, 0.f};
    const float NEG = -1e30f;
    const float scale = 0.125f;  // 1/sqrt(64)
    const unsigned ldsK = lds_off_of(&lK[0]);
    const unsigned ldsV = lds_off_of(&lV[0]);

    // ---- Q fragments straight from bf16 global (exact A-operand layout) ----
    BFrag aq[2];
    {
        int row = r0 + (lane & 15);
        const unsigned short* qp = qkv + (size_t)(b * ATT_T + row) * ATT_C3 + h * ATT_HD;
        int kb = (lane < 16) ? 0 : 8;
#pragma unroll
        for (int half = 0; half < 2; ++half) {
            aq[half].q[0] = *(const uint4*)(qp + half * 32 + kb);
            aq[half].q[1] = *(const uint4*)(qp + half * 32 + 16 + kb);
        }
    }

    float mrow[8], lrow[8];
    v8f od[4];
#pragma unroll
    for (int v = 0; v < 8; ++v) { mrow[v] = NEG; lrow[v] = 0.f; }
#pragma unroll
    for (int f = 0; f < 4; ++f) od[f] = vzero;

    int lo = r0 - 128; if (lo < 0) lo = 0;
    const int lo32 = lo & ~31;
    const int nLocal = (r0 + 16 - lo32 + 31) / 32;
    const int nChunks = nLocal + ((r0 >= 128) ? 1 : 0);
    const int rhalf = (lane & 16) ? 8 : 0;

    for (int ch = 0; ch < nChunks; ++ch) {
        const bool strided = (ch == nLocal);
        const int cs = lo32 + ch * 32;

        // ---- async-stage 32 key rows of K and V (128B each) ----
        {
            int myc = strided ? (lane * 64) : (cs + lane);
            int cc  = myc < ATT_T ? myc : ATT_T - 1;
            unsigned kgo = (unsigned)(((size_t)(b * ATT_T + cc) * ATT_C3 + ATT_C + h * ATT_HD) * 2);
            unsigned vgo = kgo + (unsigned)(ATT_C * 2);
            unsigned kl  = ldsK + (unsigned)(lane * ATT_HD) * 2;
            unsigned vl  = ldsV + (unsigned)(lane * ATT_HD) * 2;
#pragma unroll
            for (int ci = 0; ci < 8; ++ci) {
                async_ld_b128(kl + ci * 16, kgo + ci * 16, qkv);
                async_ld_b128(vl + ci * 16, vgo + ci * 16, qkv);
            }
            wait_async0();
        }
        __syncthreads();

        // ---- S(16x32) = Q . K^T : two 16x16 C tiles, 4 WMMAs ----
        v8f s[2];
#pragma unroll
        for (int tn = 0; tn < 2; ++tn) {
            int slot = tn * 16 + (lane & 15);
            int kk   = (lane < 16) ? 0 : 8;
            BFrag bk0, bk1;
            bk0.q[0] = *(const uint4*)&lK[slot * ATT_HD + kk];
            bk0.q[1] = *(const uint4*)&lK[slot * ATT_HD + kk + 16];
            bk1.q[0] = *(const uint4*)&lK[slot * ATT_HD + 32 + kk];
            bk1.q[1] = *(const uint4*)&lK[slot * ATT_HD + 32 + kk + 16];
            v8f t = wmma_bf16(aq[0], bk0, vzero);
            s[tn]  = wmma_bf16(aq[1], bk1, t);
        }

        // ---- exact sparse mask + scale; chunk row-max ----
        float pm[8];
#pragma unroll
        for (int v = 0; v < 8; ++v) pm[v] = NEG;
#pragma unroll
        for (int tn = 0; tn < 2; ++tn) {
            int slot = tn * 16 + (lane & 15);
            int c = strided ? slot * 64 : cs + slot;
#pragma unroll
            for (int v = 0; v < 8; ++v) {
                int r = r0 + v + rhalf;
                bool ok = strided ? (c <= r && (r - c) > 128)
                                  : (c <= r && (r - c) <= 128);
                float val = ok ? s[tn][v] * scale : NEG;
                s[tn][v] = val;
                pm[v] = fmaxf(pm[v], val);
            }
        }
#pragma unroll
        for (int m = 8; m >= 1; m >>= 1)
#pragma unroll
            for (int v = 0; v < 8; ++v)
                pm[v] = fmaxf(pm[v], __shfl_xor(pm[v], m, 32));

        float mnew[8], alpha[8], ps[8];
#pragma unroll
        for (int v = 0; v < 8; ++v) {
            mnew[v]  = fmaxf(mrow[v], pm[v]);
            alpha[v] = __expf(mrow[v] - mnew[v]);
            mrow[v]  = mnew[v];
            ps[v]    = 0.f;
        }

        // ---- P = exp(S-m) -> LDS in A-operand row-major ----
#pragma unroll
        for (int tn = 0; tn < 2; ++tn) {
            int col = tn * 16 + (lane & 15);
#pragma unroll
            for (int v = 0; v < 8; ++v) {
                float sv = s[tn][v];
                float p  = (sv > -1e29f) ? __expf(sv - mnew[v]) : 0.f;
                ps[v] += p;
                lP[(v + rhalf) * 32 + col] = f2bf(p);
            }
        }
#pragma unroll
        for (int m = 8; m >= 1; m >>= 1)
#pragma unroll
            for (int v = 0; v < 8; ++v)
                ps[v] += __shfl_xor(ps[v], m, 32);
#pragma unroll
        for (int v = 0; v < 8; ++v) lrow[v] = lrow[v] * alpha[v] + ps[v];
#pragma unroll
        for (int f = 0; f < 4; ++f)
#pragma unroll
            for (int v = 0; v < 8; ++v) od[f][v] *= alpha[v];
        __syncthreads();

        // ---- O += P(16x32) . V(32x64) : 4 WMMAs ----
        BFrag ap;
        {
            int row = lane & 15;
            int kb  = (lane < 16) ? 0 : 8;
            ap.q[0] = *(const uint4*)&lP[row * 32 + kb];
            ap.q[1] = *(const uint4*)&lP[row * 32 + kb + 16];
        }
#pragma unroll
        for (int f = 0; f < 4; ++f) {
            BFrag bv;
            int d  = f * 16 + (lane & 15);
            int kk = (lane < 16) ? 0 : 8;
#pragma unroll
            for (int j = 0; j < 16; ++j)
                bv.u[j] = lV[(kk + (j < 8 ? j : j + 8)) * ATT_HD + d];
            od[f] = wmma_bf16(ap, bv, od[f]);
        }
        __syncthreads();
    }

    // ---- epilogue: O / l, bf16 scatter to (B*T, C) ----
#pragma unroll
    for (int f = 0; f < 4; ++f)
#pragma unroll
        for (int v = 0; v < 8; ++v) {
            float denom = lrow[v];
            float val = (denom > 0.f) ? od[f][v] / denom : 0.f;
            int r = r0 + v + rhalf;
            out[(size_t)(b * ATT_T + r) * ATT_C + h * ATT_HD + f * 16 + (lane & 15)] = f2bf(val);
        }
}

// ---------------------------------------------------------------------------
// Launch
// ---------------------------------------------------------------------------
extern "C" void kernel_launch(void* const* d_in, const int* in_sizes, int n_in,
                              void* d_out, int out_size, void* d_ws, size_t ws_size,
                              hipStream_t stream) {
    (void)in_sizes; (void)n_in; (void)out_size; (void)ws_size;
    const float* x     = (const float*)d_in[0];   // (4096, 1024)
    const float* w_qkv = (const float*)d_in[1];   // (3072, 1024)
    const float* w_out = (const float*)d_in[2];   // (1024, 1024)
    float* outp = (float*)d_out;                  // (4096, 1024) f32

    const int M = 2 * 2048;  // 4096 token rows
    unsigned short* ws = (unsigned short*)d_ws;
    unsigned short* qkv_bf  = ws;                                   // 4096x3072
    unsigned short* attn_bf = qkv_bf  + (size_t)M * ATT_C3;         // 4096x1024
    unsigned short* x_bf    = attn_bf + (size_t)M * ATT_C;          // 4096x1024
    unsigned short* wqkv_bf = x_bf    + (size_t)M * ATT_C;          // 3072x1024
    unsigned short* wout_bf = wqkv_bf + (size_t)ATT_C3 * ATT_C;     // 1024x1024

    // 0) one-shot f32->bf16 conversions (hoists all cvt VALU out of GEMM loops)
    cvt_f32_bf16<<<dim3((M * ATT_C / 8 + 255) / 256), 256, 0, stream>>>(x, x_bf, M * ATT_C / 8);
    cvt_f32_bf16<<<dim3((ATT_C3 * ATT_C / 8 + 255) / 256), 256, 0, stream>>>(w_qkv, wqkv_bf, ATT_C3 * ATT_C / 8);
    cvt_f32_bf16<<<dim3((ATT_C * ATT_C / 8 + 255) / 256), 256, 0, stream>>>(w_out, wout_bf, ATT_C * ATT_C / 8);

    // 1) QKV projection (bf16 out): qkv = x @ w_qkv^T
    gemm_bf16<true><<<dim3(M / GBM, ATT_C3 / GBN), 256, 0, stream>>>(
        x_bf, wqkv_bf, qkv_bf, M, ATT_C3, ATT_C);

    // 2) Sparse attention per (b, h, 16-query tile)
    sparse_attn<<<dim3(2 * ATT_H * (ATT_T / 16)), 32, 0, stream>>>(qkv_bf, attn_bf);

    // 3) Output projection (f32 out): out = attn @ w_out^T
    gemm_bf16<false><<<dim3(M / GBM, ATT_C / GBN), 256, 0, stream>>>(
        attn_bf, wout_bf, outp, M, ATT_C, ATT_C);
}
